// COBlock_83872121356537
// MI455X (gfx1250) — matrix operand order repaired and checked by reading, not compiled, not run on hardware
//
#include <hip/hip_runtime.h>
#include <math.h>

typedef __attribute__((ext_vector_type(16))) __bf16 v16bf;
typedef __attribute__((ext_vector_type(2)))  __bf16 v2bf;
typedef __attribute__((ext_vector_type(8)))  float  v8f;
typedef __attribute__((ext_vector_type(4)))  unsigned int v4u;

#define DIMC  256
#define HEADS 8
#define DHEAD 64
#define INNER 512
#define NB    8
#define NN    4096
#define NM    256
#define ATT_SCALE 0.125f
#define LN_EPS 1e-5f

__device__ __forceinline__ v8f wmma_bf16(v16bf a, v16bf b, v8f c) {
  return __builtin_amdgcn_wmma_f32_16x16x32_bf16(false, a, false, b, (short)0, c,
                                                 false, false);
}

union Frag { v16bf bf; v4u u[2]; };

// A fragment 16x32 (MxK), row-major (k fast, stride ld between m rows).
// Per CDNA5 ISA 7.12.2: lane holds k = hi*8+{0..7} and 16+hi*8+{0..7} of row
// m = lane&15 -> two contiguous 16B loads.
__device__ __forceinline__ v16bf load_a16(const __bf16* __restrict__ base,
                                          int ld, int lane) {
  const int m = lane & 15, hi = (lane >> 4) & 1;
  const __bf16* p = base + (size_t)m * ld + hi * 8;
  Frag f;
  f.u[0] = *(const v4u*)(p);
  f.u[1] = *(const v4u*)(p + 16);
  return f.bf;
}

// B fragment 32x16 (KxN) from a k-fast source (stride ld between n rows).
// Lane holds k = kh*8+{0..7} and 16+kh*8+{0..7} of column n = lane&15 -> two contiguous 16B.
__device__ __forceinline__ v16bf load_bT(const __bf16* __restrict__ base,
                                         int ld, int lane) {
  const int n = lane & 15, kh = (lane >> 4) & 1;
  const __bf16* p = base + (size_t)n * ld + kh * 8;
  Frag f;
  f.u[0] = *(const v4u*)(p);
  f.u[1] = *(const v4u*)(p + 16);
  return f.bf;
}

// Butterfly reductions within a 16-lane half (one C-layout row group).
__device__ __forceinline__ float half_sum(float v) {
#pragma unroll
  for (int m = 1; m < 16; m <<= 1) v += __shfl_xor(v, m, 16);
  return v;
}
__device__ __forceinline__ float half_max(float v) {
#pragma unroll
  for (int m = 1; m < 16; m <<= 1) v = fmaxf(v, __shfl_xor(v, m, 16));
  return v;
}

// ---------------------------------------------------------------------------
// One-shot prep: transpose + convert weight W[K,OUT] (fp32) -> Wt[OUT,K] bf16
// (k-fast rows so every GEMM B-fragment is two contiguous 16B loads).
// ---------------------------------------------------------------------------
__global__ __launch_bounds__(256) void prep_weight(
    const float* __restrict__ W, __bf16* __restrict__ Wt, int K, int OUT) {
  int pid = blockIdx.x * 256 + threadIdx.x;      // (K/2)*OUT pairs
  int kp = (pid / OUT) * 2;
  int n  = pid % OUT;
  v2bf pk;
  pk[0] = (__bf16)W[(size_t)kp * OUT + n];
  pk[1] = (__bf16)W[(size_t)(kp + 1) * OUT + n];
  *(v2bf*)&Wt[(size_t)n * K + kp] = pk;
}

// One-shot prep of GEMM A inputs to bf16. MODE 0: copy A0; 1: A0+A1;
// 2: A0 + A1 where A1 is [B, C, T] (pos, channel-major) and A0 is [B*T, C].
template <int MODE>
__global__ __launch_bounds__(256) void prep_input(
    const float* __restrict__ A0, const float* __restrict__ A1, int T,
    __bf16* __restrict__ Out) {
  int pe = blockIdx.x * 256 + threadIdx.x;       // pairs over [R, 256]
  int r = pe >> 7, cp = (pe & 127) * 2;
  float a0 = A0[(size_t)r * DIMC + cp];
  float a1 = A0[(size_t)r * DIMC + cp + 1];
  if (MODE == 1) {
    a0 += A1[(size_t)r * DIMC + cp];
    a1 += A1[(size_t)r * DIMC + cp + 1];
  } else if (MODE == 2) {
    int b = r / T, t = r % T;
    a0 += A1[((size_t)b * DIMC + cp) * T + t];
    a1 += A1[((size_t)b * DIMC + cp + 1) * T + t];
  }
  v2bf pk; pk[0] = (__bf16)a0; pk[1] = (__bf16)a1;
  *(v2bf*)&Out[(size_t)r * DIMC + cp] = pk;
}

// ---------------------------------------------------------------------------
// Projection GEMM (no LDS): Out[r, 0:512] = A[r,:] @ W, K=256, A bf16
// row-major, Wt bf16 [512,256] k-fast. TRANSOUT=1 stores head-transposed
// [B,H,64,T] (for V matrices). Block 256 thr / 8 waves, 128 rows x 64 cols.
// ---------------------------------------------------------------------------
template <int TRANSOUT>
__global__ __launch_bounds__(256) void proj_kernel(
    const __bf16* __restrict__ A, const __bf16* __restrict__ Wt, int T,
    __bf16* __restrict__ Out) {
  const int tid = threadIdx.x, lane = tid & 31, w = tid >> 5;
  const int r0 = blockIdx.x * 128 + w * 16;
  const int c0 = blockIdx.y * 64;
  v8f acc[4] = {};
#pragma unroll
  for (int j = 0; j < 8; ++j) {  // K=256, step 32
    v16bf af = load_a16(A + (size_t)r0 * DIMC + j * 32, DIMC, lane);
#pragma unroll
    for (int t = 0; t < 4; ++t) {
      v16bf bf = load_bT(Wt + (size_t)(c0 + t * 16) * DIMC + j * 32, DIMC, lane);
      acc[t] = wmma_bf16(af, bf, acc[t]);
    }
  }
  const int hi = (lane >> 4) & 1, col = lane & 15;
#pragma unroll
  for (int t = 0; t < 4; ++t)
#pragma unroll
    for (int r = 0; r < 8; ++r) {
      int row = r0 + hi * 8 + r;
      if (TRANSOUT) {
        int b = row / T, tr = row % T;
        int h = blockIdx.y, d = t * 16 + col;
        Out[(((size_t)b * HEADS + h) * DHEAD + d) * T + tr] = (__bf16)acc[t][r];
      } else {
        Out[(size_t)row * INNER + c0 + t * 16 + col] = (__bf16)acc[t][r];
      }
    }
}

// ---------------------------------------------------------------------------
// Row attention (softmax over M): wave owns 16 query rows of one (b,h).
// S (16x256) in 16 C tiles; softmax in-register; P bounced via wave-local LDS
// into A-layout; out = (P @ V2)/rowsum -> AOf (bf16 merged [B,N,INNER]).
// ---------------------------------------------------------------------------
__global__ __launch_bounds__(128) void attn_feat_kernel(
    const __bf16* __restrict__ Q, const __bf16* __restrict__ K,
    const __bf16* __restrict__ V2t, __bf16* __restrict__ AOf) {
  __shared__ __align__(16) __bf16 Pl[4 * 16 * 256];
  const int tid = threadIdx.x, lane = tid & 31, w = tid >> 5;
  const int h = blockIdx.y, b = blockIdx.z;
  const int n0 = blockIdx.x * 64 + w * 16;
  const __bf16* Qb = Q + ((size_t)b * NN + n0) * INNER + h * DHEAD;
  const __bf16* Kb = K + ((size_t)b * NM) * INNER + h * DHEAD;
  const __bf16* Vb = V2t + ((size_t)b * HEADS + h) * DHEAD * NM;
  const int hi = (lane >> 4) & 1, col = lane & 15;
  __bf16* Pw = Pl + w * 16 * 256;

  v16bf qa[2];
#pragma unroll
  for (int j = 0; j < 2; ++j) qa[j] = load_a16(Qb + j * 32, INNER, lane);

  v8f s[16];
#pragma unroll
  for (int t = 0; t < 16; ++t) {
    v8f c = {};
#pragma unroll
    for (int j = 0; j < 2; ++j) {
      v16bf bf = load_bT(Kb + (size_t)t * 16 * INNER + j * 32, INNER, lane);
      c = wmma_bf16(qa[j], bf, c);
    }
    s[t] = c;
  }

  float rsum[8];
#pragma unroll
  for (int r = 0; r < 8; ++r) {
    float mx = -1e30f;
#pragma unroll
    for (int t = 0; t < 16; ++t) mx = fmaxf(mx, s[t][r]);
    mx = half_max(mx) * ATT_SCALE;
    float sm = 0.f;
#pragma unroll
    for (int t = 0; t < 16; ++t) {
      float p = __expf(s[t][r] * ATT_SCALE - mx);
      sm += p;
      Pw[(hi * 8 + r) * 256 + t * 16 + col] = (__bf16)p;
    }
    rsum[r] = half_sum(sm);
  }

  v8f o[4] = {};
#pragma unroll
  for (int j = 0; j < 8; ++j) {
    v16bf pa = load_a16(Pw + j * 32, 256, lane);
#pragma unroll
    for (int dt = 0; dt < 4; ++dt) {
      v16bf bf = load_bT(Vb + (size_t)dt * 16 * NM + j * 32, NM, lane);
      o[dt] = wmma_bf16(pa, bf, o[dt]);
    }
  }
#pragma unroll
  for (int dt = 0; dt < 4; ++dt)
#pragma unroll
    for (int r = 0; r < 8; ++r) {
      size_t row = (size_t)b * NN + n0 + hi * 8 + r;
      AOf[row * INNER + h * DHEAD + dt * 16 + col] = (__bf16)(o[dt][r] / rsum[r]);
    }
}

// ---------------------------------------------------------------------------
// Column attention (softmax over N): wave owns 16 center rows of one (b,h),
// flash-streams n in chunks of 32 with online softmax; V1 head-transposed
// [B,H,64,N]. -> AOc (bf16 merged [B,M,INNER]).
// ---------------------------------------------------------------------------
__global__ __launch_bounds__(128) void attn_center_kernel(
    const __bf16* __restrict__ Q, const __bf16* __restrict__ K,
    const __bf16* __restrict__ V1t, __bf16* __restrict__ AOc) {
  __shared__ __align__(16) __bf16 Pl[4 * 16 * 32];
  const int tid = threadIdx.x, lane = tid & 31, w = tid >> 5;
  const int task = blockIdx.x * 4 + w;  // 1024 tasks = B*H*(M/16)
  const int b = task >> 7;
  const int h = (task >> 4) & 7;
  const int m0 = (task & 15) << 4;
  const __bf16* Kb = K + ((size_t)b * NM + m0) * INNER + h * DHEAD;
  const __bf16* Qb = Q + ((size_t)b * NN) * INNER + h * DHEAD;
  const __bf16* Vb = V1t + ((size_t)b * HEADS + h) * DHEAD * NN;
  const int hi = (lane >> 4) & 1, col = lane & 15;
  __bf16* Pw = Pl + w * 16 * 32;

  v16bf ka[2];
#pragma unroll
  for (int j = 0; j < 2; ++j) ka[j] = load_a16(Kb + j * 32, INNER, lane);

  float rmax[8], rsum[8];
#pragma unroll
  for (int r = 0; r < 8; ++r) { rmax[r] = -1e30f; rsum[r] = 0.f; }
  v8f o[4] = {};

  for (int nc = 0; nc < NN / 32; ++nc) {
    const int n0 = nc * 32;
    if (nc + 1 < NN / 32)  // gfx1250 speculative prefetch of next Q chunk
      __builtin_prefetch(Qb + (size_t)(n0 + 32 + lane) * INNER, 0, 0);
    v8f st[2];
#pragma unroll
    for (int u = 0; u < 2; ++u) {
      v8f c = {};
#pragma unroll
      for (int j = 0; j < 2; ++j) {
        v16bf bf = load_bT(Qb + (size_t)(n0 + u * 16) * INNER + j * 32, INNER, lane);
        c = wmma_bf16(ka[j], bf, c);
      }
      st[u] = c;
    }
#pragma unroll
    for (int r = 0; r < 8; ++r) {
      float mx = fmaxf(st[0][r], st[1][r]) * ATT_SCALE;
      mx = half_max(mx);
      float nm = fmaxf(rmax[r], mx);
      float alpha = __expf(rmax[r] - nm);
      rmax[r] = nm;
      float p0 = __expf(st[0][r] * ATT_SCALE - nm);
      float p1 = __expf(st[1][r] * ATT_SCALE - nm);
      rsum[r] = rsum[r] * alpha + half_sum(p0 + p1);
#pragma unroll
      for (int dt = 0; dt < 4; ++dt) o[dt][r] *= alpha;
      Pw[(hi * 8 + r) * 32 + col] = (__bf16)p0;
      Pw[(hi * 8 + r) * 32 + 16 + col] = (__bf16)p1;
    }
    v16bf pa = load_a16(Pw, 32, lane);
#pragma unroll
    for (int dt = 0; dt < 4; ++dt) {
      v16bf bf = load_bT(Vb + (size_t)dt * 16 * NN + n0, NN, lane);
      o[dt] = wmma_bf16(pa, bf, o[dt]);
    }
  }
#pragma unroll
  for (int dt = 0; dt < 4; ++dt)
#pragma unroll
    for (int r = 0; r < 8; ++r) {
      size_t row = (size_t)b * NM + m0 + hi * 8 + r;
      AOc[row * INNER + h * DHEAD + dt * 16 + col] = (__bf16)(o[dt][r] / rsum[r]);
    }
}

// ---------------------------------------------------------------------------
// Output projection + residual + LayerNorm (no LDS): X = AO@Wo + Res, LN ->
// fp32 to Xf (d_out scratch) and bf16 to Xb. K=512; Wot is [256,512] k-fast.
// Wave owns a 16-row x 256-col stripe so LN stays wave-local.
// ---------------------------------------------------------------------------
__global__ __launch_bounds__(256) void out_proj_kernel(
    const __bf16* __restrict__ AO, const __bf16* __restrict__ Wot,
    const float* __restrict__ Res, const float* __restrict__ lnw,
    const float* __restrict__ lnb, float* __restrict__ Xf,
    __bf16* __restrict__ Xb) {
  const int tid = threadIdx.x, lane = tid & 31, w = tid >> 5;
  const int r0 = blockIdx.x * 128 + w * 16;
  v8f acc[16] = {};
#pragma unroll
  for (int j = 0; j < 16; ++j) {  // K=512
    v16bf af = load_a16(AO + (size_t)r0 * INNER + j * 32, INNER, lane);
#pragma unroll
    for (int t = 0; t < 16; ++t) {
      v16bf bf = load_bT(Wot + (size_t)t * 16 * INNER + j * 32, INNER, lane);
      acc[t] = wmma_bf16(af, bf, acc[t]);
    }
  }
  const int hi = (lane >> 4) & 1, col = lane & 15;
#pragma unroll
  for (int t = 0; t < 16; ++t)
#pragma unroll
    for (int r = 0; r < 8; ++r)
      acc[t][r] += Res[(size_t)(r0 + hi * 8 + r) * DIMC + t * 16 + col];
#pragma unroll
  for (int r = 0; r < 8; ++r) {
    float sm = 0.f, sq = 0.f;
#pragma unroll
    for (int t = 0; t < 16; ++t) { float x = acc[t][r]; sm += x; sq += x * x; }
    sm = half_sum(sm); sq = half_sum(sq);
    float mu = sm * (1.0f / DIMC);
    float var = sq * (1.0f / DIMC) - mu * mu;
    float rstd = rsqrtf(var + LN_EPS);
#pragma unroll
    for (int t = 0; t < 16; ++t) {
      int c = t * 16 + col;
      float y = (acc[t][r] - mu) * rstd * lnw[c] + lnb[c];
      size_t row = (size_t)(r0 + hi * 8 + r);
      Xf[row * DIMC + c] = y;
      Xb[row * DIMC + c] = (__bf16)y;
    }
  }
}

// ---------------------------------------------------------------------------
// FFN: Y = relu(X1 @ W1) @ W2 + b2 + X1 (in-place on Xf, X1 bf16 from Xb).
// Weights pre-transposed [256,256] k-fast; only H needs a wave-local LDS
// bounce into A-layout. Block 128 thr (4 waves), 64 rows.
// ---------------------------------------------------------------------------
__global__ __launch_bounds__(128) void ffn_kernel(
    const __bf16* __restrict__ Xb, const __bf16* __restrict__ W1t,
    const __bf16* __restrict__ W2t, const float* __restrict__ b2,
    float* __restrict__ Xf) {
  __shared__ __align__(16) __bf16 Hb[4 * 16 * 256];
  const int tid = threadIdx.x, lane = tid & 31, w = tid >> 5;
  const int r0 = blockIdx.x * 64 + w * 16;
  const int hi = (lane >> 4) & 1, col = lane & 15;
  __bf16* Hw = Hb + w * 16 * 256;
  {
    v8f acc[16] = {};
#pragma unroll
    for (int j = 0; j < 8; ++j) {  // K=256
      v16bf af = load_a16(Xb + (size_t)r0 * DIMC + j * 32, DIMC, lane);
#pragma unroll
      for (int t = 0; t < 16; ++t) {
        v16bf bf = load_bT(W1t + (size_t)t * 16 * DIMC + j * 32, DIMC, lane);
        acc[t] = wmma_bf16(af, bf, acc[t]);
      }
    }
#pragma unroll
    for (int t = 0; t < 16; ++t)
#pragma unroll
      for (int r = 0; r < 8; ++r)
        Hw[(hi * 8 + r) * 256 + t * 16 + col] = (__bf16)fmaxf(acc[t][r], 0.f);
  }
  {
    v8f acc[16] = {};
#pragma unroll
    for (int j = 0; j < 8; ++j) {
      v16bf af = load_a16(Hw + j * 32, 256, lane);
#pragma unroll
      for (int t = 0; t < 16; ++t) {
        v16bf bf = load_bT(W2t + (size_t)t * 16 * DIMC + j * 32, DIMC, lane);
        acc[t] = wmma_bf16(af, bf, acc[t]);
      }
    }
#pragma unroll
    for (int t = 0; t < 16; ++t)
#pragma unroll
      for (int r = 0; r < 8; ++r) {
        size_t row = (size_t)(r0 + hi * 8 + r);
        int c = t * 16 + col;
        Xf[row * DIMC + c] = acc[t][r] + b2[c] + Xf[row * DIMC + c];
      }
  }
}

// ---------------------------------------------------------------------------
extern "C" void kernel_launch(void* const* d_in, const int* in_sizes, int n_in,
                              void* d_out, int out_size, void* d_ws, size_t ws_size,
                              hipStream_t stream) {
  (void)in_sizes; (void)n_in; (void)out_size; (void)ws_size;
  const float* feat   = (const float*)d_in[0];
  const float* center = (const float*)d_in[1];
  const float* pos    = (const float*)d_in[2];
  const float* cpos   = (const float*)d_in[3];
  const float* Wq     = (const float*)d_in[4];
  const float* Wk     = (const float*)d_in[5];
  const float* Wv1    = (const float*)d_in[6];
  const float* Wv2    = (const float*)d_in[7];
  const float* Wo1    = (const float*)d_in[8];
  const float* Wo2    = (const float*)d_in[9];
  const float* ln1w   = (const float*)d_in[10];
  const float* ln1b   = (const float*)d_in[11];
  const float* ln2w   = (const float*)d_in[12];
  const float* ln2b   = (const float*)d_in[13];
  const float* f1W1   = (const float*)d_in[14];
  const float* f1W2   = (const float*)d_in[15];
  const float* f1b2   = (const float*)d_in[16];
  const float* f2W1   = (const float*)d_in[17];
  const float* f2W2   = (const float*)d_in[18];
  const float* f2b2   = (const float*)d_in[19];

  __bf16* wsb = (__bf16*)d_ws;
  size_t o = 0;
  // bf16 inputs
  __bf16* qin  = wsb + o; o += (size_t)NB * NN * DIMC;   // feat + pos
  __bf16* kin  = wsb + o; o += (size_t)NB * NM * DIMC;   // center + cpos
  __bf16* fb   = wsb + o; o += (size_t)NB * NN * DIMC;   // feat
  __bf16* cb   = wsb + o; o += (size_t)NB * NM * DIMC;   // center
  // bf16 transposed (k-fast) weights
  __bf16* Wqt  = wsb + o; o += (size_t)INNER * DIMC;
  __bf16* Wkt  = wsb + o; o += (size_t)INNER * DIMC;
  __bf16* Wv1t = wsb + o; o += (size_t)INNER * DIMC;
  __bf16* Wv2t = wsb + o; o += (size_t)INNER * DIMC;
  __bf16* Wo1t = wsb + o; o += (size_t)DIMC * INNER;
  __bf16* Wo2t = wsb + o; o += (size_t)DIMC * INNER;
  __bf16* W11t = wsb + o; o += (size_t)DIMC * DIMC;
  __bf16* W12t = wsb + o; o += (size_t)DIMC * DIMC;
  __bf16* W21t = wsb + o; o += (size_t)DIMC * DIMC;
  __bf16* W22t = wsb + o; o += (size_t)DIMC * DIMC;
  // intermediates
  __bf16* Qb   = wsb + o; o += (size_t)NB * NN * INNER;  // [B,N,512]
  __bf16* Kb   = wsb + o; o += (size_t)NB * NM * INNER;  // [B,M,512]
  __bf16* V1t  = wsb + o; o += (size_t)NB * NN * INNER;  // [B,H,64,N]
  __bf16* V2t  = wsb + o; o += (size_t)NB * NM * INNER;  // [B,H,64,M]
  __bf16* AOf  = wsb + o; o += (size_t)NB * NN * INNER;
  __bf16* AOc  = wsb + o; o += (size_t)NB * NM * INNER;
  __bf16* X1fb = wsb + o; o += (size_t)NB * NN * DIMC;
  __bf16* X1cb = wsb + o; o += (size_t)NB * NM * DIMC;

  float* outF = (float*)d_out;
  float* outC = outF + (size_t)NB * NN * DIMC;

  // --- one-shot preps (inputs + weights to bf16, weights transposed) ---
  prep_input<2><<<dim3(16384), 256, 0, stream>>>(feat,   pos,  NN, qin);
  prep_input<1><<<dim3(1024),  256, 0, stream>>>(center, cpos, NM, kin);
  prep_input<0><<<dim3(16384), 256, 0, stream>>>(feat,   nullptr, NN, fb);
  prep_input<0><<<dim3(1024),  256, 0, stream>>>(center, nullptr, NM, cb);
  prep_weight<<<dim3(256), 256, 0, stream>>>(Wq,  Wqt,  DIMC, INNER);
  prep_weight<<<dim3(256), 256, 0, stream>>>(Wk,  Wkt,  DIMC, INNER);
  prep_weight<<<dim3(256), 256, 0, stream>>>(Wv1, Wv1t, DIMC, INNER);
  prep_weight<<<dim3(256), 256, 0, stream>>>(Wv2, Wv2t, DIMC, INNER);
  prep_weight<<<dim3(256), 256, 0, stream>>>(Wo1, Wo1t, INNER, DIMC);
  prep_weight<<<dim3(256), 256, 0, stream>>>(Wo2, Wo2t, INNER, DIMC);
  prep_weight<<<dim3(128), 256, 0, stream>>>(f1W1, W11t, DIMC, DIMC);
  prep_weight<<<dim3(128), 256, 0, stream>>>(f1W2, W12t, DIMC, DIMC);
  prep_weight<<<dim3(128), 256, 0, stream>>>(f2W1, W21t, DIMC, DIMC);
  prep_weight<<<dim3(128), 256, 0, stream>>>(f2W2, W22t, DIMC, DIMC);

  // --- projections ---
  proj_kernel<0><<<dim3(256, 8), 256, 0, stream>>>(qin, Wqt,  NN, Qb);
  proj_kernel<0><<<dim3(16,  8), 256, 0, stream>>>(kin, Wkt,  NM, Kb);
  proj_kernel<1><<<dim3(256, 8), 256, 0, stream>>>(fb,  Wv1t, NN, V1t);
  proj_kernel<1><<<dim3(16,  8), 256, 0, stream>>>(cb,  Wv2t, NM, V2t);

  // --- attention ---
  attn_feat_kernel<<<dim3(64, HEADS, NB), 128, 0, stream>>>(Qb, Kb, V2t, AOf);
  attn_center_kernel<<<dim3(256), 128, 0, stream>>>(Qb, Kb, V1t, AOc);

  // --- output projection + residual + LayerNorm ---
  out_proj_kernel<<<dim3(256), 256, 0, stream>>>(AOf, Wo1t, feat,   ln1w, ln1b, outF, X1fb);
  out_proj_kernel<<<dim3(16),  256, 0, stream>>>(AOc, Wo2t, center, ln2w, ln2b, outC, X1cb);

  // --- FFN (in-place on d_out regions) ---
  ffn_kernel<<<dim3(512), 128, 0, stream>>>(X1fb, W11t, W12t, f1b2, outF);
  ffn_kernel<<<dim3(32),  128, 0, stream>>>(X1cb, W21t, W22t, f2b2, outC);
}
